// ChemROAR_64072322122026
// MI455X (gfx1250) — matrix operand order repaired and checked by reading, not compiled, not run on hardware
//
#include <hip/hip_runtime.h>
#include <hip/hip_bf16.h>
#include <math.h>

typedef __attribute__((ext_vector_type(16))) __bf16 v16bf;
typedef __attribute__((ext_vector_type(8)))  __bf16 v8bf;
typedef __attribute__((ext_vector_type(8)))  float  v8f;
typedef __attribute__((ext_vector_type(4)))  unsigned int u32x4;
typedef __attribute__((ext_vector_type(8)))  unsigned int u32x8;

#define Bz 8
#define Tz 1024
#define Dz 512
#define Hz 8
#define HDz 64
#define DFFz 1024
#define BTz (Bz*Tz)

// ---------------------------------------------------------------- WMMA core
__device__ __forceinline__ v8f wmma_bf16(v16bf a, v16bf b, v8f c) {
  // (neg_a, A, neg_b, B, c_mod, C, reuse_a, reuse_b)
  return __builtin_amdgcn_wmma_f32_16x16x32_bf16(false, a, false, b, (short)0, c,
                                                 false, false);
}

__device__ __forceinline__ v16bf join8(v8bf lo, v8bf hi) {
  return __builtin_shufflevector(lo, hi, 0,1,2,3,4,5,6,7,8,9,10,11,12,13,14,15);
}

// ------------------------------------------------ TDM: stage 2D bf16 tile
// Builds a Tensor DMA Descriptor (D#) per CDNA5 ISA and
// issues tensor_load_to_lds (inline asm = portable across toolchains).
// Loads a [rows x k_elems] bf16 tile (row stride = k_elems) into LDS.
__device__ __forceinline__ void tdm_load_tile_bf16(const void* gsrc,
                                                   unsigned lds_byte_addr,
                                                   unsigned k_elems,
                                                   unsigned rows) {
  const unsigned long long ga = (unsigned long long)(size_t)gsrc;
  u32x4 g0;
  g0[0] = 1u;                                   // count=1, user mode, no gather
  g0[1] = lds_byte_addr;                        // lds_addr [63:32]
  g0[2] = (unsigned)(ga & 0xFFFFFFFFu);         // global_addr lo
  g0[3] = (unsigned)((ga >> 32) & 0x01FFFFFFu)  // global_addr [56:32]
          | (2u << 30);                         // type = 2 ("image")
  u32x8 g1;
  g1[0] = (1u << 16);                           // data_size=1 (2 bytes); mask=0
  g1[1] = (k_elems & 0xFFFFu) << 16;            // tensor_dim0 lo16 @ bits[63:48]
  g1[2] = ((k_elems >> 16) & 0xFFFFu)           // tensor_dim0 hi16
          | ((rows & 0xFFFFu) << 16);           // tensor_dim1 lo16
  g1[3] = ((rows >> 16) & 0xFFFFu)              // tensor_dim1 hi16
          | (k_elems << 16);                    // tile_dim0 @ bits[127:112]
  g1[4] = rows;                                 // tile_dim1 (tile_dim2=0)
  g1[5] = k_elems;                              // tensor_dim0_stride lo32
  g1[6] = 0u;                                   // stride hi / dim1_stride
  g1[7] = 0u;
  u32x4 gz = {0u, 0u, 0u, 0u};                  // groups 2/3: zero (2D tile)
  asm volatile("tensor_load_to_lds %0, %1, %2, %3"
               :: "s"(g0), "s"(g1), "s"(gz), "s"(gz)
               : "memory");
}

// One wave computes a 16x16 f32 tile: A from LDS (staged by TDM), Bt from
// global. Bt row-major "transposed B": Bt[n][k] = B[k][n].
__device__ __forceinline__ v8f wmma_tile_ldsA(const __bf16* lA, int lda,
                                              const __bf16* __restrict__ Bt,
                                              int ldb, int K, int lane) {
  const int mr = lane & 15, kh = lane >> 4;
  const __bf16* arow = lA + (size_t)mr * lda;  // A-frag: row m = lane&15 (LDS)
  const __bf16* brow = Bt + (size_t)mr * ldb;  // B-frag: col n = lane&15
  v8f acc = {};
  for (int k = 0; k < K; k += 32) {
    // A-fragment: K = 16*kh + e, contiguous 16 bf16 (two 8-chunks)
    v8bf a0 = *(const v8bf*)(arow + k + 16 * kh);
    v8bf a1 = *(const v8bf*)(arow + k + 16 * kh + 8);
    // B-fragment: K = 16*kh + e, contiguous 16 bf16
    v16bf bv = *(const v16bf*)(brow + k + 16 * kh);
    if (k + 32 < K) __builtin_prefetch(brow + k + 32, 0, 1);  // global_prefetch_b8
    acc = wmma_bf16(join8(a0, a1), bv, acc);
  }
  return acc;
}

// Pure-global variant (unused; flash attention loads Q fragments directly)
__device__ __forceinline__ v8f wmma_tile(const __bf16* __restrict__ A, int lda,
                                         const __bf16* __restrict__ Bt, int ldb,
                                         int K, int lane) {
  const int mr = lane & 15, kh = lane >> 4;
  const __bf16* arow = A + (size_t)mr * lda;
  const __bf16* brow = Bt + (size_t)mr * ldb;
  v8f acc = {};
  for (int k = 0; k < K; k += 32) {
    v8bf a0 = *(const v8bf*)(arow + k + 8 * kh);
    v8bf a1 = *(const v8bf*)(arow + k + 16 + 8 * kh);
    v16bf bv = *(const v16bf*)(brow + k + 16 * kh);
    acc = wmma_bf16(join8(a0, a1), bv, acc);
  }
  return acc;
}

// ------------------------------------------------------- weight transposes
__global__ void k_transpose_bf16(const float* __restrict__ src,
                                 __bf16* __restrict__ dst, int rows, int cols) {
  int idx = blockIdx.x * blockDim.x + threadIdx.x;
  if (idx >= rows * cols) return;
  int r = idx / cols, c = idx - r * cols;
  dst[(size_t)c * rows + r] = (__bf16)src[idx];
}

// ----------------------------------------------------------- layernorm(bf16)
// one wave per row of D=512 (16 elems/lane), shfl_xor reduction
__global__ void k_layernorm_bf16(const float* __restrict__ x,
                                 const float* __restrict__ g,
                                 const float* __restrict__ b,
                                 __bf16* __restrict__ out) {
  const int lane = threadIdx.x & 31, wid = threadIdx.x >> 5;
  const int row = blockIdx.x * 8 + wid;
  const float* xr = x + (size_t)row * Dz;
  float v[16];
  float s = 0.f;
#pragma unroll
  for (int i = 0; i < 16; ++i) { v[i] = xr[lane * 16 + i]; s += v[i]; }
#pragma unroll
  for (int m = 16; m >= 1; m >>= 1) s += __shfl_xor(s, m, 32);
  const float mean = s * (1.0f / Dz);
  float var = 0.f;
#pragma unroll
  for (int i = 0; i < 16; ++i) { float d = v[i] - mean; var += d * d; }
#pragma unroll
  for (int m = 16; m >= 1; m >>= 1) var += __shfl_xor(var, m, 32);
  const float rstd = rsqrtf(var * (1.0f / Dz) + 1e-5f);
#pragma unroll
  for (int i = 0; i < 16; ++i) {
    int d = lane * 16 + i;
    out[(size_t)row * Dz + d] = (__bf16)((v[i] - mean) * rstd * g[d] + b[d]);
  }
}

// ------------------------------------------------------------ QKV GEMM (f32)
// qkv[8192 x 1536] = h1 @ W_attn ; Wt = W_attn^T bf16 [1536][512]
// A-tile (16x512 bf16, 16KB) staged once per block via TDM.
__global__ void k_gemm_qkv(const __bf16* __restrict__ h,
                           const __bf16* __restrict__ Wt,
                           float* __restrict__ qkv) {
  __shared__ __bf16 atile[16 * Dz];
  const int lane = threadIdx.x & 31, wid = threadIdx.x >> 5;
  const int row0 = blockIdx.x * 16;
  const int col0 = (blockIdx.y * 8 + wid) * 16;
  if (wid == 0) {
    tdm_load_tile_bf16(h + (size_t)row0 * Dz, (unsigned)(size_t)atile, Dz, 16);
    __builtin_amdgcn_s_wait_tensorcnt(0);
  }
  __syncthreads();
  v8f c = wmma_tile_ldsA(atile, Dz, Wt + (size_t)col0 * Dz, Dz, Dz, lane);
  const int n = lane & 15, half = lane >> 4;
#pragma unroll
  for (int r = 0; r < 8; ++r)
    qkv[(size_t)(row0 + r + 8 * half) * (3 * Dz) + col0 + n] = c[r];
}

// ------------------------------------------- embed add + RoPE + head packing
// Q,K: bf16 [B,H,T,64] ; Vt: bf16 [B,H,64,T]
__global__ void k_rope_pack(const float* __restrict__ qkv,
                            const int* __restrict__ x_type,
                            const float* __restrict__ seq_order,
                            const float* __restrict__ type_emb,
                            __bf16* __restrict__ Q, __bf16* __restrict__ K,
                            __bf16* __restrict__ Vt) {
  const int idx = blockIdx.x * blockDim.x + threadIdx.x;  // < 8*1024*8*32
  const int p = idx & 31;            // dim pair (2p,2p+1) within head
  const int h = (idx >> 5) & 7;
  const int t = (idx >> 8) & 1023;
  const int b = idx >> 18;
  const size_t row = (size_t)b * Tz + t;
  const int d0 = h * HDz + 2 * p;
  float q0 = qkv[row * 1536 + d0],        q1 = qkv[row * 1536 + d0 + 1];
  float k0 = qkv[row * 1536 + 512 + d0],  k1 = qkv[row * 1536 + 512 + d0 + 1];
  float v0 = qkv[row * 1536 + 1024 + d0], v1 = qkv[row * 1536 + 1024 + d0 + 1];
  const int tq = x_type[(size_t)b * (Tz + 1) + t];
  const int tk = x_type[(size_t)b * (Tz + 1) + t + 1];
  q0 += type_emb[(size_t)tq * 1024 + d0];
  q1 += type_emb[(size_t)tq * 1024 + d0 + 1];
  k0 += type_emb[(size_t)tk * 1024 + 512 + d0];
  k1 += type_emb[(size_t)tk * 1024 + 512 + d0 + 1];
  if (p < 16) {  // RoPE on first DPR=32 dims, interleaved pairs, freq_p
    const float freq = __expf(-(float)(2 * p) * (9.2103403719761836f / 32.0f));
    const float aq = seq_order[(size_t)b * (Tz + 1) + t] * freq;
    const float ak = seq_order[(size_t)b * (Tz + 1) + t + 1] * freq;
    float cq = __cosf(aq), sq = __sinf(aq);
    float ck = __cosf(ak), sk = __sinf(ak);
    float nq0 = q0 * cq - q1 * sq, nq1 = q1 * cq + q0 * sq;
    float nk0 = k0 * ck - k1 * sk, nk1 = k1 * ck + k0 * sk;
    q0 = nq0; q1 = nq1; k0 = nk0; k1 = nk1;
  }
  const size_t bh = (size_t)b * Hz + h;
  const size_t qi = (bh * Tz + t) * HDz + 2 * p;
  Q[qi] = (__bf16)q0; Q[qi + 1] = (__bf16)q1;
  K[qi] = (__bf16)k0; K[qi + 1] = (__bf16)k1;
  Vt[(bh * HDz + 2 * p) * Tz + t]     = (__bf16)v0;
  Vt[(bh * HDz + 2 * p + 1) * Tz + t] = (__bf16)v1;
}

// ------------------------------------------------------ flash attention
// one wave = 16 q rows of one (b,h); xout = x_value + attn_out  (f32 [B,T,D])
__global__ void k_flash_attn(const __bf16* __restrict__ Q,
                             const __bf16* __restrict__ Kc,
                             const __bf16* __restrict__ Vt,
                             const float* __restrict__ xv,
                             float* __restrict__ xout) {
  __shared__ __bf16 plds[8][512];  // per-wave 16x32 P staging tile
  const int lane = threadIdx.x & 31, wid = threadIdx.x >> 5;
  const int wg = blockIdx.x * 8 + wid;
  const int qt = wg & 63, hh = (wg >> 6) & 7, bb = wg >> 9;
  const int q0 = qt * 16;
  const size_t bh = (size_t)bb * Hz + hh;
  const __bf16* Qb = Q + bh * Tz * HDz;
  const __bf16* Kb = Kc + bh * Tz * HDz;
  const __bf16* Vb = Vt + bh * HDz * Tz;
  const int mr = lane & 15, kh = lane >> 4;

  // hoist Q A-fragments (d-steps 0..31 and 32..63)
  v16bf qa[2];
#pragma unroll
  for (int dstp = 0; dstp < 2; ++dstp) {
    const __bf16* qr = Qb + (size_t)(q0 + mr) * HDz + 32 * dstp + 16 * kh;
    qa[dstp] = join8(*(const v8bf*)qr, *(const v8bf*)(qr + 8));
  }

  v8f o[4] = {};           // output accum, 4 dv tiles of 16
  float mrow[8], lrow[8];
#pragma unroll
  for (int r = 0; r < 8; ++r) { mrow[r] = -3.0e38f; lrow[r] = 0.f; }

  const int nkb = (q0 + 16 + 31) >> 5;  // 32-key blocks covering keys < q0+16
  for (int kb = 0; kb < nkb; ++kb) {
    const int key0 = kb * 32;
    // ---- S = Q @ K^T : two 16-key sub-tiles, two d-steps each
    v8f s[2] = {};
#pragma unroll
    for (int sub = 0; sub < 2; ++sub) {
#pragma unroll
      for (int dstp = 0; dstp < 2; ++dstp) {
        const __bf16* kr =
            Kb + (size_t)(key0 + 16 * sub + mr) * HDz + 32 * dstp + 16 * kh;
        s[sub] = wmma_bf16(qa[dstp], *(const v16bf*)kr, s[sub]);
      }
    }
    // ---- scale + causal mask + running row max
    float mc[8];
#pragma unroll
    for (int r = 0; r < 8; ++r) {
      const int mg = q0 + r + 8 * kh;  // this lane's rows
      float s0 = s[0][r] * 0.125f;     // 1/sqrt(64)
      float s1 = s[1][r] * 0.125f;
      if (key0 + mr > mg) s0 = -1.0e30f;
      if (key0 + 16 + mr > mg) s1 = -1.0e30f;
      s[0][r] = s0; s[1][r] = s1;
      mc[r] = fmaxf(s0, s1);
    }
#pragma unroll
    for (int msk = 8; msk >= 1; msk >>= 1)
#pragma unroll
      for (int r = 0; r < 8; ++r) mc[r] = fmaxf(mc[r], __shfl_xor(mc[r], msk, 32));
    // ---- online softmax update; stage P as bf16 A-operand in LDS
    float alpha[8], rs[8];
#pragma unroll
    for (int r = 0; r < 8; ++r) {
      const float mn = fmaxf(mrow[r], mc[r]);
      alpha[r] = __expf(mrow[r] - mn);
      mrow[r] = mn;
      const float p0 = __expf(s[0][r] - mn);
      const float p1 = __expf(s[1][r] - mn);
      plds[wid][(r + 8 * kh) * 32 + mr]      = (__bf16)p0;
      plds[wid][(r + 8 * kh) * 32 + 16 + mr] = (__bf16)p1;
      rs[r] = p0 + p1;
    }
#pragma unroll
    for (int msk = 8; msk >= 1; msk >>= 1)
#pragma unroll
      for (int r = 0; r < 8; ++r) rs[r] += __shfl_xor(rs[r], msk, 32);
#pragma unroll
    for (int r = 0; r < 8; ++r) lrow[r] = lrow[r] * alpha[r] + rs[r];
#pragma unroll
    for (int dvt = 0; dvt < 4; ++dvt)
#pragma unroll
      for (int r = 0; r < 8; ++r) o[dvt][r] *= alpha[r];
    // ---- P A-fragment from LDS (row m, K = 16*kh + e contiguous)
    v16bf pa = join8(*(const v8bf*)&plds[wid][mr * 32 + 16 * kh],
                     *(const v8bf*)&plds[wid][mr * 32 + 16 * kh + 8]);
    // ---- O += P @ V : V^T rows are contiguous over keys
#pragma unroll
    for (int dvt = 0; dvt < 4; ++dvt) {
      const __bf16* vr = Vb + (size_t)(16 * dvt + mr) * Tz + key0 + 16 * kh;
      o[dvt] = wmma_bf16(pa, *(const v16bf*)vr, o[dvt]);
    }
  }
  // ---- epilogue: out/l + residual with x_value
  float inv[8];
#pragma unroll
  for (int r = 0; r < 8; ++r) inv[r] = 1.0f / lrow[r];
#pragma unroll
  for (int dvt = 0; dvt < 4; ++dvt)
#pragma unroll
    for (int r = 0; r < 8; ++r) {
      const int t = q0 + r + 8 * kh;
      const int col = hh * HDz + dvt * 16 + mr;
      const size_t idx = ((size_t)bb * Tz + t) * Dz + col;
      xout[idx] = xv[idx] + o[dvt][r] * inv[r];
    }
}

// -------------------------------------------------------- FFN1 + SwiGLU
// mid[8192 x 1024] = silu(h2@W1g + b1g) * (h2@W1a + b1a), bf16
__global__ void k_ffn1(const __bf16* __restrict__ h2,
                       const __bf16* __restrict__ W1t,
                       const float* __restrict__ b1,
                       __bf16* __restrict__ mid) {
  __shared__ __bf16 atile[16 * Dz];
  const int lane = threadIdx.x & 31, wid = threadIdx.x >> 5;
  const int row0 = blockIdx.x * 16;
  const int col0 = (blockIdx.y * 8 + wid) * 16;
  if (wid == 0) {
    tdm_load_tile_bf16(h2 + (size_t)row0 * Dz, (unsigned)(size_t)atile, Dz, 16);
    __builtin_amdgcn_s_wait_tensorcnt(0);
  }
  __syncthreads();
  v8f ca = wmma_tile_ldsA(atile, Dz, W1t + (size_t)col0 * Dz, Dz, Dz, lane);
  v8f cg = wmma_tile_ldsA(atile, Dz, W1t + (size_t)(col0 + DFFz) * Dz, Dz, Dz, lane);
  const int n = lane & 15, half = lane >> 4;
  const float ba = b1[col0 + n], bg = b1[DFFz + col0 + n];
#pragma unroll
  for (int r = 0; r < 8; ++r) {
    const float a = ca[r] + ba;
    const float g = cg[r] + bg;
    const float sg = g / (1.0f + __expf(-g));  // silu
    mid[(size_t)(row0 + r + 8 * half) * DFFz + col0 + n] = (__bf16)(sg * a);
  }
}

// -------------------------------------------------- FFN2 + final residual
__global__ void k_ffn2(const __bf16* __restrict__ mid,
                       const __bf16* __restrict__ W2t,
                       const float* __restrict__ b2,
                       const float* __restrict__ xres,
                       float* __restrict__ out) {
  __shared__ __bf16 atile[16 * DFFz];
  const int lane = threadIdx.x & 31, wid = threadIdx.x >> 5;
  const int row0 = blockIdx.x * 16;
  const int col0 = (blockIdx.y * 8 + wid) * 16;
  if (wid == 0) {
    tdm_load_tile_bf16(mid + (size_t)row0 * DFFz, (unsigned)(size_t)atile, DFFz, 16);
    __builtin_amdgcn_s_wait_tensorcnt(0);
  }
  __syncthreads();
  v8f c = wmma_tile_ldsA(atile, DFFz, W2t + (size_t)col0 * DFFz, DFFz, DFFz, lane);
  const int n = lane & 15, half = lane >> 4;
  const float bb = b2[col0 + n];
#pragma unroll
  for (int r = 0; r < 8; ++r) {
    const size_t idx = (size_t)(row0 + r + 8 * half) * Dz + col0 + n;
    out[idx] = xres[idx] + c[r] + bb;
  }
}

// --------------------------------------------------------------- launcher
extern "C" void kernel_launch(void* const* d_in, const int* in_sizes, int n_in,
                              void* d_out, int out_size, void* d_ws, size_t ws_size,
                              hipStream_t stream) {
  (void)in_sizes; (void)n_in; (void)out_size; (void)ws_size;
  const int*   x_type    = (const int*)d_in[0];
  const float* x_value   = (const float*)d_in[1];
  const float* seq_order = (const float*)d_in[2];
  const float* W_attn    = (const float*)d_in[3];
  const float* type_emb  = (const float*)d_in[4];
  const float* ln1_g = (const float*)d_in[5];
  const float* ln1_b = (const float*)d_in[6];
  const float* ln2_g = (const float*)d_in[7];
  const float* ln2_b = (const float*)d_in[8];
  const float* W1 = (const float*)d_in[9];
  const float* b1 = (const float*)d_in[10];
  const float* W2 = (const float*)d_in[11];
  const float* b2 = (const float*)d_in[12];
  float* out = (float*)d_out;

  char* ws = (char*)d_ws;
  size_t off = 0;
  auto take = [&](size_t bytes) {
    char* p = ws + off;
    off += (bytes + 255) & ~(size_t)255;
    return p;
  };
  __bf16* WattnT = (__bf16*)take((size_t)1536 * 512 * 2);
  __bf16* W1T    = (__bf16*)take((size_t)2048 * 512 * 2);
  __bf16* W2T    = (__bf16*)take((size_t)512 * 1024 * 2);
  __bf16* h1     = (__bf16*)take((size_t)BTz * Dz * 2);
  float*  qkv    = (float*) take((size_t)BTz * 1536 * 4);
  __bf16* Qb     = (__bf16*)take((size_t)BTz * Dz * 2);
  __bf16* Kb     = (__bf16*)take((size_t)BTz * Dz * 2);
  __bf16* Vtb    = (__bf16*)take((size_t)BTz * Dz * 2);
  float*  xbuf   = (float*) take((size_t)BTz * Dz * 4);
  __bf16* h2     = (__bf16*)take((size_t)BTz * Dz * 2);
  __bf16* mid    = (__bf16*)take((size_t)BTz * DFFz * 2);

  const dim3 blk(256);
  k_transpose_bf16<<<(512 * 1536 + 255) / 256, blk, 0, stream>>>(W_attn, WattnT, 512, 1536);
  k_transpose_bf16<<<(512 * 2048 + 255) / 256, blk, 0, stream>>>(W1, W1T, 512, 2048);
  k_transpose_bf16<<<(1024 * 512 + 255) / 256, blk, 0, stream>>>(W2, W2T, 1024, 512);
  k_layernorm_bf16<<<BTz / 8, blk, 0, stream>>>(x_value, ln1_g, ln1_b, h1);
  k_gemm_qkv<<<dim3(BTz / 16, 12), blk, 0, stream>>>(h1, WattnT, qkv);
  k_rope_pack<<<(Bz * Tz * Hz * 32) / 256, blk, 0, stream>>>(qkv, x_type, seq_order,
                                                             type_emb, Qb, Kb, Vtb);
  k_flash_attn<<<(Bz * Hz * (Tz / 16)) / 8, blk, 0, stream>>>(Qb, Kb, Vtb, x_value, xbuf);
  k_layernorm_bf16<<<BTz / 8, blk, 0, stream>>>(xbuf, ln2_g, ln2_b, h2);
  k_ffn1<<<dim3(BTz / 16, DFFz / 128), blk, 0, stream>>>(h2, W1T, b1, mid);
  k_ffn2<<<dim3(BTz / 16, Dz / 128), blk, 0, stream>>>(mid, W2T, b2, xbuf, out);
}
